// GhostLinear_54855322304539
// MI455X (gfx1250) — matrix operand order repaired and checked by reading, not compiled, not run on hardware
//
#include <hip/hip_runtime.h>

typedef __attribute__((ext_vector_type(16))) _Float16 v16h;
typedef __attribute__((ext_vector_type(8)))  _Float16 v8h;
typedef __attribute__((ext_vector_type(4)))  _Float16 v4h;
typedef __attribute__((ext_vector_type(8)))  float    v8f;
typedef __attribute__((ext_vector_type(4)))  float    v4f;
typedef __attribute__((ext_vector_type(4)))  int      v4i;

constexpr int T    = 64;     // tokens (M)
constexpr int K    = 4096;   // in features
constexpr int O    = 11008;  // out features
constexpr int KT   = 128;    // K tile per half staged in LDS
constexpr int XPAD = 8;      // f16 pad per xs row -> row stride 272B (breaks 64-bank stride)
constexpr int KSPLIT = 2;    // K halves per block (waves 4..7 take the upper half)
constexpr int KH   = K / KSPLIT;        // 2048
constexpr int OPW  = 16;     // out features per wave
constexpr int OPB  = 4 * OPW;           // 64 out features per block
constexpr int THREADS = 256;            // 8 wave32: 4 N-waves x 2 K-halves

__global__ __launch_bounds__(THREADS)
void ghost_linear_wmma(const float* __restrict__ x,
                       const int*   __restrict__ gidx,
                       const float* __restrict__ lut,
                       const float* __restrict__ scale,
                       float* __restrict__ out)
{
    __shared__ float    lutS[256];
    __shared__ _Float16 xs[KSPLIT][T][KT + XPAD];   // ~34.8 KB
    __shared__ float    redS[4][4][32][8];          // 16 KB: kh=1 partials

    const int tid  = threadIdx.x;
    const int lane = tid & 31;
    const int wave = tid >> 5;
    const int nw   = wave & 3;    // N-tile wave id
    const int kh   = wave >> 2;   // K half
    const int col  = lane & 15;   // N column (B/C) or M row (A) selector
    const int hi   = lane >> 4;   // half-wave: selects contiguous K sub-range k = 16*hi + s

    // LUT -> LDS (256 threads, 1 entry each)
    if (tid < 256) lutS[tid] = lut[tid];

    const int   o  = blockIdx.x * OPB + nw * OPW + col;  // this lane's out-feature row
    const float sc = scale[o];
    const int* __restrict__ irow = gidx + (size_t)o * K + kh * KH;

    v8f acc[4] = {};  // 4 M-tiles of 16x16 f32 (partial over this K half)

    for (int it = 0; it < KH / KT; ++it) {   // 16 iterations
        const int kk0 = it * KT;

        // Prefetch next index chunk for this lane's row (global_prefetch_b8)
        if (kk0 + KT < KH)
            __builtin_prefetch(irow + kk0 + KT + hi * 16, 0, 1);

        // ---- Stage BOTH halves' x tiles [64 x 128] f32 -> f16 (coalesced b128 loads)
        #pragma unroll
        for (int i = 0; i < (KSPLIT * T * KT / 4) / THREADS; ++i) {  // 16
            const int v   = i * THREADS + tid;
            const int t   = v >> 11;          // which K half (2048 float4 each)
            const int rem = v & 2047;
            const int row = rem >> 5;         // 32 float4 per row
            const int kq  = (rem & 31) << 2;
            v4f xv = *(const v4f*)(x + (size_t)row * K + t * KH + kk0 + kq);
            *(v4h*)&xs[t][row][kq] = __builtin_convertvector(xv, v4h);
        }
        __syncthreads();

        #pragma unroll
        for (int k2 = 0; k2 < KT; k2 += 32) {
            // ---- B fragment: 16 indices -> LDS LUT gather -> scale -> f16
            const int* ip = irow + kk0 + k2 + hi * 16;
            v4i iv0 = *(const v4i*)(ip + 0);
            v4i iv1 = *(const v4i*)(ip + 4);
            v4i iv2 = *(const v4i*)(ip + 8);
            v4i iv3 = *(const v4i*)(ip + 12);
            v16h b;
            #pragma unroll
            for (int e = 0; e < 4; ++e) {
                b[0  + e] = (_Float16)(lutS[iv0[e]] * sc);
                b[4  + e] = (_Float16)(lutS[iv1[e]] * sc);
                b[8  + e] = (_Float16)(lutS[iv2[e]] * sc);
                b[12 + e] = (_Float16)(lutS[iv3[e]] * sc);
            }

            // ---- A fragments (4 M-tiles) from this half's LDS tile + WMMA
            const int kb = k2 + hi * 16;
            #pragma unroll
            for (int mt = 0; mt < 4; ++mt) {
                const int row = mt * 16 + col;
                v8h alo = *(const v8h*)&xs[kh][row][kb];
                v8h ahi = *(const v8h*)&xs[kh][row][kb + 8];
                v16h a  = __builtin_shufflevector(alo, ahi,
                                                  0,1,2,3,4,5,6,7,8,9,10,11,12,13,14,15);
                acc[mt] = __builtin_amdgcn_wmma_f32_16x16x32_f16(
                    false, a, false, b, (short)0, acc[mt], false, false);
            }
        }
        __syncthreads();
    }

    // ---- Deterministic cross-K-half reduction via LDS, then store.
    if (kh == 1) {
        #pragma unroll
        for (int mt = 0; mt < 4; ++mt)
            *(v8f*)&redS[nw][mt][lane][0] = acc[mt];
    }
    __syncthreads();
    if (kh == 0) {
        #pragma unroll
        for (int mt = 0; mt < 4; ++mt) {
            v8f other = *(const v8f*)&redS[nw][mt][lane][0];
            v8f sum   = acc[mt] + other;
            #pragma unroll
            for (int r = 0; r < 8; ++r) {
                const int tok = mt * 16 + hi * 8 + r;   // C/D layout: VGPR r -> M = r + 8*hi
                out[(size_t)tok * O + o] = sum[r];
            }
        }
    }
}

extern "C" void kernel_launch(void* const* d_in, const int* in_sizes, int n_in,
                              void* d_out, int out_size, void* d_ws, size_t ws_size,
                              hipStream_t stream) {
    const float* x     = (const float*)d_in[0];
    const int*   gidx  = (const int*)d_in[1];
    const float* lut   = (const float*)d_in[2];
    const float* scale = (const float*)d_in[3];
    float*       out   = (float*)d_out;

    dim3 grid(O / OPB);     // 172 blocks
    dim3 block(THREADS);    // 8 wave32 (4 N-tiles x 2 K-halves)
    ghost_linear_wmma<<<grid, block, 0, stream>>>(x, gidx, lut, scale, out);
}
